// MultiHeadAttention_2267742732467
// MI455X (gfx1250) — matrix operand order repaired and checked by reading, not compiled, not run on hardware
//
#include <hip/hip_runtime.h>

typedef __attribute__((ext_vector_type(16))) __bf16 v16bf;
typedef __attribute__((ext_vector_type(8)))  __bf16 v8bf;
typedef __attribute__((ext_vector_type(4)))  __bf16 v4bf;
typedef __attribute__((ext_vector_type(8)))  float  v8f;

static __device__ __forceinline__ v8f wmma_bf16(v16bf a, v16bf b, v8f c) {
  // D(f32 16x16) = A(bf16 16x32) * B(bf16 32x16) + C
  return __builtin_amdgcn_wmma_f32_16x16x32_bf16(false, a, false, b, (short)0, c,
                                                 false, false);
}

static __device__ __forceinline__ v16bf pack16(v8bf lo, v8bf hi) {
  v16bf r;
#pragma unroll
  for (int i = 0; i < 8; ++i) { r[i] = lo[i]; r[i + 8] = hi[i]; }
  return r;
}

static __device__ __forceinline__ v8f zero8() {
  v8f r;
#pragma unroll
  for (int i = 0; i < 8; ++i) r[i] = 0.0f;
  return r;
}

// --- DPP16 butterfly reductions over a 16-lane half (no DS traffic) --------
template <int CTRL>
static __device__ __forceinline__ float fdpp(float x) {
  return __int_as_float(
      __builtin_amdgcn_mov_dpp(__float_as_int(x), CTRL, 0xF, 0xF, true));
}
static __device__ __forceinline__ float red_max16(float v) {
  v = fmaxf(v, fdpp<0xB1>(v));    // quad_perm [1,0,3,2]  (xor 1)
  v = fmaxf(v, fdpp<0x4E>(v));    // quad_perm [2,3,0,1]  (xor 2)
  v = fmaxf(v, fdpp<0x141>(v));   // row_half_mirror      (pairs quads in 8)
  v = fmaxf(v, fdpp<0x140>(v));   // row_mirror           (pairs 8-groups in 16)
  return v;
}
static __device__ __forceinline__ float red_sum16(float v) {
  v += fdpp<0xB1>(v);
  v += fdpp<0x4E>(v);
  v += fdpp<0x141>(v);
  v += fdpp<0x140>(v);
  return v;
}

// Low 32 bits of a generic pointer to __shared__ == DS-usable LDS address.
static __device__ __forceinline__ unsigned lds_off32(const void* p) {
  return (unsigned)(size_t)p;
}

// Async DMA: 16B global -> LDS, tracked by ASYNCcnt (GVS addressing mode).
static __device__ __forceinline__ void async_b128(unsigned lds_dst,
                                                  unsigned gbyte_off,
                                                  const void* base) {
  asm volatile("global_load_async_to_lds_b128 %0, %1, %2"
               :: "v"(lds_dst), "v"(gbyte_off),
                  "s"((unsigned long long)(size_t)base)
               : "memory");
}
static __device__ __forceinline__ void wait_async_le4() {
  asm volatile("s_wait_asynccnt 0x4" ::: "memory");
}
static __device__ __forceinline__ void wait_async_0() {
  asm volatile("s_wait_asynccnt 0x0" ::: "memory");
}

// ---------------------------------------------------------------------------
// One-time fp32 -> bf16 conversion (elementwise, 8 elems/thread).
// ---------------------------------------------------------------------------
__global__ __launch_bounds__(256) void cvt_f32_bf16(
    const float* __restrict__ src, __bf16* __restrict__ dst, int n8) {
  const int i = blockIdx.x * 256 + threadIdx.x;
  if (i >= n8) return;
  float t[8];
  *(float4*)&t[0] = ((const float4*)src)[i * 2];
  *(float4*)&t[4] = ((const float4*)src)[i * 2 + 1];
  v8bf c;
#pragma unroll
  for (int j = 0; j < 8; ++j) c[j] = (__bf16)t[j];
  *(v8bf*)(dst + (size_t)i * 8) = c;
}

// ---------------------------------------------------------------------------
// One-time W[K][N] fp32 -> Wt[N][K] bf16 (32x32 LDS tile transpose).
// ---------------------------------------------------------------------------
__global__ __launch_bounds__(256) void transpose_cvt(
    const float* __restrict__ W, __bf16* __restrict__ Wt, int K, int N) {
  __shared__ float tile[32][33];
  const int k0 = blockIdx.y * 32;
  const int n0 = blockIdx.x * 32;
  const int r  = threadIdx.x >> 3;         // 0..31
  const int c4 = (threadIdx.x & 7) << 2;   // 0..28
  const float4 f = *(const float4*)&W[(size_t)(k0 + r) * N + n0 + c4];
  tile[r][c4 + 0] = f.x;
  tile[r][c4 + 1] = f.y;
  tile[r][c4 + 2] = f.z;
  tile[r][c4 + 3] = f.w;
  __syncthreads();
  v4bf o;
#pragma unroll
  for (int j = 0; j < 4; ++j) o[j] = (__bf16)tile[c4 + j][r];
  *(v4bf*)&Wt[(size_t)(n0 + r) * K + k0 + c4] = o;
}

// ---------------------------------------------------------------------------
// GEMM: Y[M,N] = X[M,K](bf16) @ Wt[N,K](bf16)^T + bias; OutT = bf16 or fp32.
// 8 waves, 128x128 tile, K-step 32. Double-buffered LDS fed entirely by
// GLOBAL_LOAD_ASYNC_TO_LDS_B128 (4 DMAs/thread/step, zero VALU staging).
// ---------------------------------------------------------------------------
#define LDS_STR 40  // 32 + 8 pad (bf16) -> conflict-free 16B fragment gathers

template <typename OutT>
__global__ __launch_bounds__(256) void gemm_bf16_nt(
    const __bf16* __restrict__ X, const __bf16* __restrict__ Wt,
    const float* __restrict__ bias, OutT* __restrict__ Y,
    int M, int N, int K) {
  __shared__ __attribute__((aligned(16))) __bf16 As[2][128 * LDS_STR];
  __shared__ __attribute__((aligned(16))) __bf16 Bs[2][128 * LDS_STR];

  const int tid  = threadIdx.x;
  const int wave = tid >> 5;
  const int lane = tid & 31;
  const int l16  = lane & 15;
  const int lhi  = lane >> 4;
  const int wm = wave >> 1;  // 0..3
  const int wn = wave & 1;   // 0..1
  const int block_m = blockIdx.y * 128;
  const int block_n = blockIdx.x * 128;

  v8f acc[2][4];
#pragma unroll
  for (int i = 0; i < 2; ++i)
#pragma unroll
    for (int j = 0; j < 4; ++j) acc[i][j] = zero8();

  // 128x32 bf16 tile = 512 x 16B chunks; 2 chunks/thread/matrix.
  auto stage = [&](int buf, int k0) {
#pragma unroll
    for (int j = 0; j < 2; ++j) {
      const int c   = tid * 2 + j;
      const int row = c >> 2;
      const int ko  = (c & 3) << 3;
      async_b128(lds_off32(&As[buf][row * LDS_STR + ko]),
                 (unsigned)(((size_t)(block_m + row) * K + k0 + ko) * 2), X);
      async_b128(lds_off32(&Bs[buf][row * LDS_STR + ko]),
                 (unsigned)(((size_t)(block_n + row) * K + k0 + ko) * 2), Wt);
    }
  };

  stage(0, 0);
  const int steps = K >> 5;
  for (int it = 0; it < steps; ++it) {
    const int cur = it & 1;
    if (it + 1 < steps) {
      stage(cur ^ 1, (it + 1) << 5);
      wait_async_le4();  // previous tile's DMAs (this wave) retired
    } else {
      wait_async_0();
    }
    __syncthreads();  // everyone's DMAs for buf[cur] retired

    v16bf afr[2], bfr[4];
#pragma unroll
    for (int mi = 0; mi < 2; ++mi) {
      const int row = wm * 32 + mi * 16 + l16;
      const int kb  = lhi << 3;  // A layout: K 0-7/16-23 vs 8-15/24-31
      afr[mi] = pack16(*(const v8bf*)&As[cur][row * LDS_STR + kb],
                       *(const v8bf*)&As[cur][row * LDS_STR + kb + 16]);
    }
#pragma unroll
    for (int ni = 0; ni < 4; ++ni) {
      const int col = wn * 64 + ni * 16 + l16;
      const int kb  = lhi << 3;  // B layout: K 0-7/16-23 vs 8-15/24-31
      bfr[ni] = pack16(*(const v8bf*)&Bs[cur][col * LDS_STR + kb],
                       *(const v8bf*)&Bs[cur][col * LDS_STR + kb + 16]);
    }
#pragma unroll
    for (int mi = 0; mi < 2; ++mi)
#pragma unroll
      for (int ni = 0; ni < 4; ++ni)
        acc[mi][ni] = wmma_bf16(afr[mi], bfr[ni], acc[mi][ni]);

    __syncthreads();  // all reads of buf[cur] done before it is re-filled
  }

#pragma unroll
  for (int mi = 0; mi < 2; ++mi) {
    const int row0 = block_m + wm * 32 + mi * 16 + lhi * 8;
#pragma unroll
    for (int ni = 0; ni < 4; ++ni) {
      const int col = block_n + wn * 64 + ni * 16 + l16;
      const float bv = bias[col];
#pragma unroll
      for (int r = 0; r < 8; ++r)
        Y[(size_t)(row0 + r) * N + col] = (OutT)(acc[mi][ni][r] + bv);
    }
  }
}

// ---------------------------------------------------------------------------
// Flash attention over bf16 qh/kh/vh. 8 waves x 16 queries, shared 64-key
// K/V tiles (async DMA for K, LDS-transposed V). One softmax update per row
// per 64-key tile; DPP butterflies; exp2-domain; bf16 ctx out.
// ---------------------------------------------------------------------------
#define S_LEN 2048
#define DM    1024
#define HD    64
#define KT    64           // keys per tile
#define AST   72           // LDS stride (64 + 8 pad)

__global__ __launch_bounds__(256) void attn_wmma(
    const __bf16* __restrict__ qh, const __bf16* __restrict__ kh,
    const __bf16* __restrict__ vh, const int* __restrict__ mask,
    __bf16* __restrict__ ctx) {
  __shared__ __attribute__((aligned(16))) __bf16 Ks[KT * AST];       // [key][depth]
  __shared__ __attribute__((aligned(16))) __bf16 Vs[HD * AST];       // [depth][key]
  __shared__ __attribute__((aligned(16))) __bf16 Ps[8 * 16 * AST];   // per-wave P

  const int tid  = threadIdx.x;
  const int wave = tid >> 5;
  const int lane = tid & 31;
  const int l16  = lane & 15;
  const int lhi  = lane >> 4;
  const int b  = blockIdx.z;
  const int h  = blockIdx.y;
  const int q0 = blockIdx.x * 128 + wave * 16;

  // Q A-fragments: two K=32 frags straight from bf16 global (16B loads).
  v16bf aq[2];
#pragma unroll
  for (int kd = 0; kd < 2; ++kd) {
    const __bf16* src =
        qh + (size_t)(b * S_LEN + q0 + l16) * DM + h * HD + kd * 32 + lhi * 8;
    aq[kd] = pack16(*(const v8bf*)src, *(const v8bf*)(src + 16));
  }

  // exp2-domain constants: logits*log2e = s*(0.125*log2e) + mask*(-1e9*log2e)
  const float SCL   = 0.125f * 1.4426950408889634f;
  const float MASKC = -1.4426950409e9f;

  float mrow[8], lrow[8];
  v8f acc[4];
#pragma unroll
  for (int r = 0; r < 8; ++r) { mrow[r] = -3.0e38f; lrow[r] = 0.0f; }
#pragma unroll
  for (int ni = 0; ni < 4; ++ni) acc[ni] = zero8();

  // staging: 64 keys x 64 depth bf16 = 512 x 16B chunks, 2 per thread
  const int c0     = tid * 2;
  const int st_k0  = c0 >> 3;          // key of first chunk
  const int st_dg0 = (c0 & 7) << 3;    // depth offset of first chunk

  for (int kt = 0; kt < S_LEN; kt += KT) {
    __syncthreads();
#pragma unroll
    for (int j = 0; j < 2; ++j) {
      const int c  = c0 + j;
      const int kl = c >> 3;
      const int dg = (c & 7) << 3;
      // K tile: async DMA straight into LDS [key][depth]
      async_b128(lds_off32(&Ks[kl * AST + dg]),
                 (unsigned)(((size_t)(b * S_LEN + kt + kl) * DM + h * HD + dg) * 2),
                 kh);
      // V tile transposed [depth][key]: 16B load + 8 b16 LDS stores
      const v8bf tv =
          *(const v8bf*)&vh[(size_t)(b * S_LEN + kt + kl) * DM + h * HD + dg];
#pragma unroll
      for (int i = 0; i < 8; ++i) Vs[(dg + i) * AST + kl] = tv[i];
    }
    wait_async_0();
    __syncthreads();

    // ---- all four 16-key score fragments first (8 WMMAs)
    v8f sfr[4];
#pragma unroll
    for (int t = 0; t < 4; ++t) {
      v8f s = zero8();
#pragma unroll
      for (int kd = 0; kd < 2; ++kd) {
        const int key = t * 16 + l16;
        const int db  = kd * 32 + lhi * 8;
        s = wmma_bf16(aq[kd],
                      pack16(*(const v8bf*)&Ks[key * AST + db],
                             *(const v8bf*)&Ks[key * AST + db + 16]),
                      s);
      }
      sfr[t] = s;
    }
    float madd[4];
#pragma unroll
    for (int t = 0; t < 4; ++t)
      madd[t] = (float)mask[b * S_LEN + kt + t * 16 + l16] * MASKC;

    // ---- one online-softmax update per row per 64-key tile
#pragma unroll
    for (int r = 0; r < 8; ++r) {
      float sv[4];
#pragma unroll
      for (int t = 0; t < 4; ++t) sv[t] = sfr[t][r] * SCL + madd[t];
      float mv = fmaxf(fmaxf(sv[0], sv[1]), fmaxf(sv[2], sv[3]));
      mv = red_max16(mv);
      const float mnew = fmaxf(mrow[r], mv);
      const float corr = __builtin_amdgcn_exp2f(mrow[r] - mnew);
      mrow[r] = mnew;
      float pe[4];
#pragma unroll
      for (int t = 0; t < 4; ++t) pe[t] = __builtin_amdgcn_exp2f(sv[t] - mnew);
      float rs = (pe[0] + pe[1]) + (pe[2] + pe[3]);
      rs = red_sum16(rs);
      lrow[r] = lrow[r] * corr + rs;
#pragma unroll
      for (int ni = 0; ni < 4; ++ni) acc[ni][r] *= corr;
#pragma unroll
      for (int t = 0; t < 4; ++t)
        Ps[wave * (16 * AST) + (lhi * 8 + r) * AST + t * 16 + l16] = (__bf16)pe[t];
    }
    __syncthreads();

    // ---- ctx += P(16x64) @ V(64x64): two K=32 chunks x 4 depth frags
#pragma unroll
    for (int kk = 0; kk < 2; ++kk) {
      const int kb = kk * 32 + (lhi << 3);
      const v16bf ap =
          pack16(*(const v8bf*)&Ps[wave * (16 * AST) + l16 * AST + kb],
                 *(const v8bf*)&Ps[wave * (16 * AST) + l16 * AST + kb + 16]);
#pragma unroll
      for (int ni = 0; ni < 4; ++ni) {
        const int dcol = ni * 16 + l16;
        const int kbv  = kk * 32 + (lhi << 3);
        acc[ni] = wmma_bf16(ap,
                            pack16(*(const v8bf*)&Vs[dcol * AST + kbv],
                                   *(const v8bf*)&Vs[dcol * AST + kbv + 16]),
                            acc[ni]);
      }
    }
  }

  // normalize, store bf16 ctx in [B, S, H*HD] layout (A-input of final GEMM)
#pragma unroll
  for (int r = 0; r < 8; ++r) {
    const float linv = 1.0f / lrow[r];
    const int qrow = q0 + lhi * 8 + r;
#pragma unroll
    for (int ni = 0; ni < 4; ++ni) {
      const int dcol = ni * 16 + l16;
      ctx[(size_t)(b * S_LEN + qrow) * DM + h * HD + dcol] =
          (__bf16)(acc[ni][r] * linv);
    }
  }
}

// ---------------------------------------------------------------------------
extern "C" void kernel_launch(void* const* d_in, const int* in_sizes, int n_in,
                              void* d_out, int out_size, void* d_ws, size_t ws_size,
                              hipStream_t stream) {
  // setup_inputs order: v, k, q, mask, Wq, bq, Wk, bk, Wv, bv, Wo, bo
  const float* v_in = (const float*)d_in[0];
  const float* k_in = (const float*)d_in[1];
  const float* q_in = (const float*)d_in[2];
  const int*   mask = (const int*)d_in[3];
  const float* Wq = (const float*)d_in[4];
  const float* bq = (const float*)d_in[5];
  const float* Wk = (const float*)d_in[6];
  const float* bk = (const float*)d_in[7];
  const float* Wv = (const float*)d_in[8];
  const float* bv = (const float*)d_in[9];
  const float* Wo = (const float*)d_in[10];
  const float* bo = (const float*)d_in[11];
  float* out = (float*)d_out;

  const size_t act_b = (size_t)8192 * 1024 * 2;  // 16 MB bf16 activation
  const size_t w_b   = (size_t)1024 * 1024 * 2;  // 2 MB bf16 weight
  char* ws = (char*)d_ws;
  __bf16* qb  = (__bf16*)(ws);                      // bf16 copies of inputs
  __bf16* kb2 = (__bf16*)(ws + act_b);
  __bf16* vb2 = (__bf16*)(ws + 2 * act_b);
  __bf16* Wqt = (__bf16*)(ws + 3 * act_b);          // transposed bf16 weights
  __bf16* Wkt = (__bf16*)(ws + 3 * act_b + w_b);
  __bf16* Wvt = (__bf16*)(ws + 3 * act_b + 2 * w_b);
  __bf16* Wot = (__bf16*)(ws + 3 * act_b + 3 * w_b);
  __bf16* qh  = (__bf16*)(ws + 3 * act_b + 4 * w_b);
  __bf16* kh  = (__bf16*)(ws + 4 * act_b + 4 * w_b);
  __bf16* vh  = (__bf16*)(ws + 5 * act_b + 4 * w_b);
  __bf16* ctx = (__bf16*)(ws + 6 * act_b + 4 * w_b);

  const dim3 blk(256);
  const int n8 = 8192 * 1024 / 8;
  cvt_f32_bf16<<<dim3(n8 / 256), blk, 0, stream>>>(q_in, qb, n8);
  cvt_f32_bf16<<<dim3(n8 / 256), blk, 0, stream>>>(k_in, kb2, n8);
  cvt_f32_bf16<<<dim3(n8 / 256), blk, 0, stream>>>(v_in, vb2, n8);
  transpose_cvt<<<dim3(32, 32), blk, 0, stream>>>(Wq, Wqt, 1024, 1024);
  transpose_cvt<<<dim3(32, 32), blk, 0, stream>>>(Wk, Wkt, 1024, 1024);
  transpose_cvt<<<dim3(32, 32), blk, 0, stream>>>(Wv, Wvt, 1024, 1024);
  transpose_cvt<<<dim3(32, 32), blk, 0, stream>>>(Wo, Wot, 1024, 1024);

  const dim3 gg(1024 / 128, 8192 / 128);  // (8, 64)
  gemm_bf16_nt<__bf16><<<gg, blk, 0, stream>>>(qb,  Wqt, bq, qh, 8192, 1024, 1024);
  gemm_bf16_nt<__bf16><<<gg, blk, 0, stream>>>(kb2, Wkt, bk, kh, 8192, 1024, 1024);
  gemm_bf16_nt<__bf16><<<gg, blk, 0, stream>>>(vb2, Wvt, bv, vh, 8192, 1024, 1024);
  attn_wmma<<<dim3(S_LEN / 128, 16, 4), blk, 0, stream>>>(qh, kh, vh, mask, ctx);
  gemm_bf16_nt<float><<<gg, blk, 0, stream>>>(ctx, Wot, bo, out, 8192, 1024, 1024);

  (void)in_sizes; (void)n_in; (void)out_size; (void)ws_size;
}
